// VectorQuantizerEMA_8022998909243
// MI455X (gfx1250) — matrix operand order repaired and checked by reading, not compiled, not run on hardware
//
#include <hip/hip_runtime.h>
#include <hip/hip_bf16.h>

typedef __attribute__((ext_vector_type(16))) __bf16 v16bf;
typedef __attribute__((ext_vector_type(8)))  float  v8f;

#define K_CODES 1024
#define CDIM    256
#define SPAT    4096              // 16*16*16
#define BATCH   16
#define NVEC    (BATCH * SPAT)    // 65536 vectors
#define ZQ_ELEMS (BATCH * CDIM * SPAT)  // 16777216

#define ROWS 64  // rows (vectors) per block in the argmin GEMM -> 4x codebook reuse
#define AP 264   // LDS A-tile pitch in bf16 elements (256 + 8 pad, 16B-aligned rows)
#define GP 257   // LDS gather-tile pitch in floats (odd stride -> conflict-free)

__device__ __forceinline__ unsigned short f32_to_bf16_rne(float f) {
    unsigned int u = __float_as_uint(f);
    unsigned int r = u + 0x7FFFu + ((u >> 16) & 1u);
    return (unsigned short)(r >> 16);
}

// Shared-predicate running argmin update: one v_cmp + two v_cndmask,
// no fminf canonicalization (v_max x,x) traffic.
__device__ __forceinline__ void argmin_upd(float s, int code, float& best, int& bidx) {
    const bool lt = s < best;
    best = lt ? s : best;
    bidx = lt ? code : bidx;
}

// ---------------------------------------------------------------------------
// Kernel 0: codebook f32 -> bf16, plus per-code squared norms (f32).
// ---------------------------------------------------------------------------
__global__ void __launch_bounds__(256)
vq_prep(const float* __restrict__ cb, unsigned short* __restrict__ cbb,
        float* __restrict__ norms) {
    const int k = blockIdx.x;
    const int t = threadIdx.x;
    float v = cb[k * CDIM + t];
    cbb[k * CDIM + t] = f32_to_bf16_rne(v);
    float sq = v * v;
    for (int off = 16; off >= 1; off >>= 1) sq += __shfl_xor(sq, off, 32);
    __shared__ float red[8];
    if ((t & 31) == 0) red[t >> 5] = sq;
    __syncthreads();
    if (t == 0) {
        float s = 0.f;
        for (int w = 0; w < 8; ++w) s += red[w];
        norms[k] = s;
    }
}

// ---------------------------------------------------------------------------
// Kernel 1: fused transpose + bf16 WMMA distance GEMM + argmin.
// Block: 64 vectors (4 m-tiles) x 1024 codes; 8 waves split 64 code tiles.
// kc-outer / mt-inner: one B fragment feeds 4 WMMAs (4 acc banks) -> B read
// once per code tile (4x less codebook L2 traffic). The compiler hoists the
// loop-invariant A fragments into high VGPRs (wave32 MSB addressing).
// Branchless running argmin (strict < keeps first occurrence per lane).
// ---------------------------------------------------------------------------
__global__ void __launch_bounds__(256)
vq_argmin(const float* __restrict__ ze, const unsigned short* __restrict__ cbb,
          const float* __restrict__ norms, int* __restrict__ idxOut) {
    __shared__ __align__(16) unsigned short As[ROWS * AP];  // 33792 B
    __shared__ float redV[8][ROWS];
    __shared__ int   redI[8][ROWS];

    const int t     = threadIdx.x;
    const int nBase = blockIdx.x * ROWS;     // 64 consecutive flat vectors (same b)
    const int b     = nBase >> 12;           // / 4096
    const int sBase = nBase & 4095;

    // Stage A tile: 64 rows x 256 channels, f32 (stride SPAT) -> bf16 LDS.
    // Two channels per thread-iteration -> packed ds_store_b32.
    {
        const int sl = t & 63;
        const int c0 = (t >> 6) * 2;         // 0,2,4,6
        const float* src = ze + ((size_t)b * CDIM) * SPAT + (size_t)(sBase + sl);
        for (int c = c0; c < CDIM; c += 8) {
            const unsigned int lo = f32_to_bf16_rne(src[(size_t)c * SPAT]);
            const unsigned int hi = f32_to_bf16_rne(src[(size_t)(c + 1) * SPAT]);
            *(unsigned int*)(As + sl * AP + c) = lo | (hi << 16);
        }
    }
    __syncthreads();

    const int lane = t & 31;
    const int wave = t >> 5;
    const int lmod = lane & 15;              // row m (A) / col n (B) within tile
    const int lh   = lane >> 4;              // lane half

    float best[4][8];
    int   bidx[4][8];
    #pragma unroll
    for (int mt = 0; mt < 4; ++mt)
        #pragma unroll
        for (int r = 0; r < 8; ++r) { best[mt][r] = 3.4e38f; bidx[mt][r] = 0x7FFFFFFF; }

    const unsigned short* arow0 = As + lmod * AP;

    // Each wave: 8 code tiles of 16 codes
    #pragma unroll 1
    for (int i = 0; i < 8; ++i) {
        const int code = (wave * 8 + i) * 16 + lmod;     // this lane's B column
        const unsigned short* brow = cbb + (size_t)code * CDIM;

        v8f acc0 = {}, acc1 = {}, acc2 = {}, acc3 = {};
        #pragma unroll
        for (int kc = 0; kc < 8; ++kc) {
            // B 32x16 bf16: lane half selects contiguous K block of 16
            union { uint4 q[2]; v16bf v; } ub;
            const int e0 = kc * 32 + lh * 16;
            ub.q[0] = *(const uint4*)(brow + e0);
            ub.q[1] = *(const uint4*)(brow + e0 + 8);

            // A 16x32 bf16: half 0 -> K {0..15}, half 1 -> K {16..31}
            const int a0 = kc * 32 + lh * 16;
            const int a1 = kc * 32 + lh * 16 + 8;
            union { uint4 q[2]; v16bf v; } ua;

            ua.q[0] = *(const uint4*)(arow0 + a0);
            ua.q[1] = *(const uint4*)(arow0 + a1);
            acc0 = __builtin_amdgcn_wmma_f32_16x16x32_bf16(false, ua.v, false, ub.v, (short)0, acc0, false, false);

            ua.q[0] = *(const uint4*)(arow0 + 16 * AP + a0);
            ua.q[1] = *(const uint4*)(arow0 + 16 * AP + a1);
            acc1 = __builtin_amdgcn_wmma_f32_16x16x32_bf16(false, ua.v, false, ub.v, (short)0, acc1, false, false);

            ua.q[0] = *(const uint4*)(arow0 + 32 * AP + a0);
            ua.q[1] = *(const uint4*)(arow0 + 32 * AP + a1);
            acc2 = __builtin_amdgcn_wmma_f32_16x16x32_bf16(false, ua.v, false, ub.v, (short)0, acc2, false, false);

            ua.q[0] = *(const uint4*)(arow0 + 48 * AP + a0);
            ua.q[1] = *(const uint4*)(arow0 + 48 * AP + a1);
            acc3 = __builtin_amdgcn_wmma_f32_16x16x32_bf16(false, ua.v, false, ub.v, (short)0, acc3, false, false);
        }

        const float nrm = norms[code];
        // Branchless running argmin. Codes increase per lane across i, so a
        // strict < preserves first-occurrence (lowest index) on exact ties.
        #pragma unroll
        for (int r = 0; r < 8; ++r) {
            argmin_upd(__builtin_fmaf(-2.f, acc0[r], nrm), code, best[0][r], bidx[0][r]);
            argmin_upd(__builtin_fmaf(-2.f, acc1[r], nrm), code, best[1][r], bidx[1][r]);
            argmin_upd(__builtin_fmaf(-2.f, acc2[r], nrm), code, best[2][r], bidx[2][r]);
            argmin_upd(__builtin_fmaf(-2.f, acc3[r], nrm), code, best[3][r], bidx[3][r]);
        }
    }

    // Reduce argmin across the 16 lanes of each half (branchless, with
    // lowest-index tie-break to match jnp.argmin).
    #pragma unroll
    for (int mt = 0; mt < 4; ++mt) {
        #pragma unroll
        for (int r = 0; r < 8; ++r) {
            float bv = best[mt][r];
            int   bi = bidx[mt][r];
            #pragma unroll
            for (int off = 8; off >= 1; off >>= 1) {
                const float ov = __shfl_xor(bv, off, 16);
                const int   oi = __shfl_xor(bi, off, 16);
                const bool take = (ov < bv) || (ov == bv && oi < bi);
                bv = take ? ov : bv;
                bi = take ? oi : bi;
            }
            best[mt][r] = bv;
            bidx[mt][r] = bi;
        }
    }
    if (lmod == 0) {
        #pragma unroll
        for (int mt = 0; mt < 4; ++mt) {
            const int mbase = mt * 16 + lh * 8;  // half 0 -> m 0..7, half 1 -> m 8..15
            #pragma unroll
            for (int r = 0; r < 8; ++r) {
                redV[wave][mbase + r] = best[mt][r];
                redI[wave][mbase + r] = bidx[mt][r];
            }
        }
    }
    __syncthreads();
    if (t < ROWS) {
        float bv = redV[0][t];
        int   bi = redI[0][t];
        for (int w = 1; w < 8; ++w) {
            const float v  = redV[w][t];
            const int   i2 = redI[w][t];
            const bool take = (v < bv) || (v == bv && i2 < bi);
            bv = take ? v  : bv;
            bi = take ? i2 : bi;
        }
        idxOut[nBase + t] = bi;
    }
}

// ---------------------------------------------------------------------------
// Kernel 2: gather codebook rows (f32), write z_q transposed to [B,C,D,H,W],
// emit indices as float, deterministic per-block loss partials.
// ---------------------------------------------------------------------------
__global__ void __launch_bounds__(256)
vq_gather(const float* __restrict__ ze, const float* __restrict__ cb,
          const int* __restrict__ idxIn, float* __restrict__ out,
          float* __restrict__ lossPart) {
    __shared__ float tile[64 * GP];
    __shared__ int   sidx[64];
    __shared__ float red[8];

    const int t     = threadIdx.x;
    const int blk   = blockIdx.x;
    const int b     = blk >> 6;
    const int sBase = (blk & 63) * 64;
    const int nBase = b * SPAT + sBase;

    if (t < 64) {
        const int id = idxIn[nBase + t];
        sidx[t] = id;
        out[ZQ_ELEMS + 1 + nBase + t] = (float)id;   // indices (as f32)
    }
    __syncthreads();

    // Gather 64 codebook rows, coalesced along C, into padded LDS tile
    for (int s = 0; s < 64; ++s)
        tile[s * GP + t] = cb[(size_t)sidx[s] * CDIM + t];
    __syncthreads();

    // Write-out coalesced along spatial dim; fuse loss accumulation
    const int ts = t & 63;
    const int tc = t >> 6;                   // 0..3
    float acc = 0.f;
    const size_t outb = ((size_t)b * CDIM) * SPAT + (size_t)(sBase + ts);
    for (int c = tc; c < CDIM; c += 4) {
        const float  v   = tile[ts * GP + c];
        const size_t off = outb + (size_t)c * SPAT;
        const float  d   = ze[off] - v;
        acc += d * d;
        out[off] = v;
    }
    for (int off = 16; off >= 1; off >>= 1) acc += __shfl_xor(acc, off, 32);
    if ((t & 31) == 0) red[t >> 5] = acc;
    __syncthreads();
    if (t == 0) {
        float s = 0.f;
        for (int w = 0; w < 8; ++w) s += red[w];
        lossPart[blk] = s;                   // deterministic: one slot per block
    }
}

// ---------------------------------------------------------------------------
// Kernel 3: deterministic fixed-order sum of partials -> vq_loss scalar
// ---------------------------------------------------------------------------
__global__ void vq_finalize(const float* __restrict__ lossPart, float* __restrict__ out) {
    if (threadIdx.x == 0 && blockIdx.x == 0) {
        float s = 0.f;
        for (int i = 0; i < 1024; ++i) s += lossPart[i];
        out[ZQ_ELEMS] = 0.5f * (s / (float)ZQ_ELEMS);
    }
}

// ---------------------------------------------------------------------------
extern "C" void kernel_launch(void* const* d_in, const int* in_sizes, int n_in,
                              void* d_out, int out_size, void* d_ws, size_t ws_size,
                              hipStream_t stream) {
    const float* z_e = (const float*)d_in[0];   // [16,256,16,16,16] f32
    const float* cb  = (const float*)d_in[1];   // [1024,256] f32
    float* out = (float*)d_out;                 // z_q (16777216) | vq_loss (1) | indices (65536)

    // workspace layout
    char* ws = (char*)d_ws;
    unsigned short* cbb      = (unsigned short*)(ws);                 // 524288 B
    float*          norms    = (float*)(ws + 524288);                 //   4096 B
    int*            idxBuf   = (int*)  (ws + 528384);                 // 262144 B
    float*          lossPart = (float*)(ws + 790528);                 //   4096 B

    vq_prep    <<<K_CODES,     256, 0, stream>>>(cb, cbb, norms);
    vq_argmin  <<<NVEC / ROWS, 256, 0, stream>>>(z_e, cbb, norms, idxBuf);
    vq_gather  <<<1024,        256, 0, stream>>>(z_e, cb, idxBuf, out, lossPart);
    vq_finalize<<<1,           64,  0, stream>>>(lossPart, out);
}
